// MoeLayer_80264348827720
// MI455X (gfx1250) — matrix operand order repaired and checked by reading, not compiled, not run on hardware
//
#include <hip/hip_runtime.h>

#define T_TOK 8192
#define DDIM  1024
#define FDIM  4096
#define NEXP  8
#define FCH   1024              // F chunk per LDS pass (4 chunks)

typedef __bf16 bf16;
typedef __bf16 bf16x16 __attribute__((ext_vector_type(16)));
typedef __bf16 bf16x4  __attribute__((ext_vector_type(4)));
typedef float  f32x8   __attribute__((ext_vector_type(8)));
typedef unsigned int u32x4 __attribute__((ext_vector_type(4)));
typedef int  i32x8 __attribute__((ext_vector_type(8)));
typedef int  i32x4 __attribute__((ext_vector_type(4)));

#if defined(__has_builtin)
#if __has_builtin(__builtin_amdgcn_tensor_load_to_lds) && __has_builtin(__builtin_amdgcn_s_wait_tensorcnt)
#define HAVE_TDM 1
#endif
#endif

// branch-free gelu (tanh form rewritten as sigmoid, native exp/rcp)
__device__ __forceinline__ float gelu_f(float x) {
    float u = -1.5957691216057308f * (x + 0.044715f * x * x * x);
    return x * __builtin_amdgcn_rcpf(1.f + __expf(u));
}

// build one A fragment (v16bf) from a row-major LDS row pointer (pairs contiguous)
__device__ __forceinline__ bf16x16 afrag_rows(const bf16* rowbase, int hs) {
    bf16x16 a;
#pragma unroll
    for (int r = 0; r < 8; ++r) {
        int ka = (r < 4) ? (hs * 8 + 2 * r) : (16 + hs * 8 + 2 * (r - 4));
        a[2 * r]     = rowbase[ka];
        a[2 * r + 1] = rowbase[ka + 1];
    }
    return a;
}

// ---------------------------------------------------------------- utilities

__global__ void k_zero_counts(int* counts) {
    if (threadIdx.x < NEXP) counts[threadIdx.x] = 0;
}

__global__ void k_convert_bf16x4(const float4* __restrict__ in,
                                 bf16* __restrict__ out, int n4) {
    int i = blockIdx.x * blockDim.x + threadIdx.x;
    if (i < n4) {
        float4 v = in[i];
        bf16x4 o;
        o[0] = (bf16)v.x; o[1] = (bf16)v.y; o[2] = (bf16)v.z; o[3] = (bf16)v.w;
        ((bf16x4*)out)[i] = o;
    }
}

// Pack W [E][K][N] fp32 into WMMA B-fragment tiles (32Kx16N), bf16.
__global__ void k_pack_b(const float* __restrict__ in, bf16* __restrict__ out,
                         int K, int N) {
    int wid  = blockIdx.x * (blockDim.x >> 5) + (threadIdx.x >> 5);
    int lane = threadIdx.x & 31;
    int kt_per = K >> 5, nt_per = N >> 4;
    int tiles_per_e = nt_per * kt_per;
    int e   = wid / tiles_per_e;
    int rem = wid - e * tiles_per_e;
    int nt  = rem / kt_per;
    int kt  = rem - nt * kt_per;
    int nlo = lane & 15, hs = lane >> 4;

    const float* src = in + ((size_t)e * K + (size_t)kt * 32) * N + nt * 16 + nlo;
    bf16x16 frag;
#pragma unroll
    for (int r = 0; r < 8; ++r) {
        int kbd = (r < 4) ? (hs * 8 + 2 * r) : (16 + hs * 8 + 2 * (r - 4));
        frag[2 * r]     = (bf16)src[(size_t)kbd * N];
        frag[2 * r + 1] = (bf16)src[(size_t)(kbd + 1) * N];
    }
    *(bf16x16*)(out + (size_t)wid * 512 + lane * 16) = frag;
}

// ---------------------------------------------------------------- gating
__global__ void k_gate_topk(const float* __restrict__ gx,
                            const float* __restrict__ Wg,
                            const float* __restrict__ bg,
                            int*   __restrict__ counts,
                            int*   __restrict__ list,
                            float* __restrict__ wlist) {
    const int wave = threadIdx.x >> 5;
    const int lane = threadIdx.x & 31;
    const int t = blockIdx.x * 8 + wave;

    float acc[NEXP];
#pragma unroll
    for (int e = 0; e < NEXP; ++e) acc[e] = 0.f;

    for (int d = lane; d < DDIM; d += 32) {
        float g = gx[(size_t)t * DDIM + d];
#pragma unroll
        for (int e = 0; e < NEXP; ++e) acc[e] += g * Wg[d * NEXP + e];
    }
#pragma unroll
    for (int e = 0; e < NEXP; ++e)
        for (int off = 16; off > 0; off >>= 1)
            acc[e] += __shfl_xor(acc[e], off, 32);

    if (lane == 0) {
        float l[NEXP];
#pragma unroll
        for (int e = 0; e < NEXP; ++e) l[e] = acc[e] + bg[e];
        int e0 = 0;
#pragma unroll
        for (int e = 1; e < NEXP; ++e) if (l[e] > l[e0]) e0 = e;
        int e1 = (e0 == 0) ? 1 : 0;
#pragma unroll
        for (int e = 0; e < NEXP; ++e) if (e != e1 && e != e0 && l[e] > l[e1]) e1 = e;
        float w0 = 1.f / (1.f + expf(l[e1] - l[e0]));
        float w1 = 1.f - w0;
        int p0 = atomicAdd(&counts[e0], 1);
        list [e0 * T_TOK + p0] = t * 2 + 0;
        wlist[e0 * T_TOK + p0] = w0;
        int p1 = atomicAdd(&counts[e1], 1);
        list [e1 * T_TOK + p1] = t * 2 + 1;
        wlist[e1 * T_TOK + p1] = w1;
    }
}

// ---------------------------------------------------------------- fused FFN
// grid=(T/32, E), block=256 (8 waves). BM=32 routed rows per block.
// LDS: x rows row-major 32x1024 bf16 (64KB, filled by TDM gather) +
//      h-chunk A-fragments 2mt x 32kt x 1KB (64KB).
// F processed in 4 chunks of 1024; y accumulated in 128 persistent VGPRs/wave.
// Every B fragment feeds 2 WMMAs (m0,m1) -> half the L2 weight traffic of BM=16.
__launch_bounds__(256)
__global__ void k_moe_ffn(const bf16*  __restrict__ xin,   // [T][D] bf16 row-major
                          const bf16*  __restrict__ w1p,   // packed B-frags
                          const float* __restrict__ b1,
                          const bf16*  __restrict__ w2p,   // packed B-frags
                          const float* __restrict__ b2,
                          const int*   __restrict__ counts,
                          const int*   __restrict__ list,
                          const float* __restrict__ wlist,
                          float* __restrict__ outp) {
    extern __shared__ char smem[];
    bf16*  xrow  = (bf16*)smem;                              // 32*1024 bf16 = 64KB
    bf16*  hfrag = (bf16*)(smem + 32 * DDIM * 2);            // 2*32*512 bf16 = 64KB
    int*   tk    = (int*)(smem + 32 * DDIM * 2 + 2 * (FCH / 32) * 512 * 2);
    float* wt    = (float*)(tk + 32);

    const int e    = blockIdx.y;
    const int cnt  = counts[e];
    const int base = blockIdx.x * 32;
    if (base >= cnt) return;                       // uniform across block
    const int tid  = threadIdx.x;

    if (tid < 32) {
        int idx = base + tid;
        int packed = (idx < cnt) ? list[e * T_TOK + idx] : -1;
        tk[tid] = packed;
        wt[tid] = (idx < cnt) ? wlist[e * T_TOK + idx] : 0.f;
    }
    __syncthreads();

    const int wv   = tid >> 5;
    const int lane = tid & 31;
    const int nlo  = lane & 15;
    const int hs   = lane >> 4;

    // ---- gather 32 routed token rows into LDS (row-major)
#if HAVE_TDM
    if (wv == 0) {
        // TDM gather-mode descriptor: 16 x 16-bit row indices per op (ISA 8.7)
        unsigned lds0 = (unsigned)(size_t)xrow;
        unsigned long long ga = (unsigned long long)(size_t)xin;
#pragma unroll
        for (int half = 0; half < 2; ++half) {
            u32x4 g0;
            g0[0] = 1u | (1u << 31);                         // count=1, gather_mode=1, 16b idx
            g0[1] = lds0 + half * 16 * DDIM * 2;             // lds_addr
            g0[2] = (unsigned)(ga & 0xFFFFFFFFull);          // global_addr[31:0]
            g0[3] = (unsigned)((ga >> 32) & 0x01FFFFFFull) | (2u << 30); // addr[56:32], type=2
            i32x8 g1;
            g1[0] = (1 << 16);                               // data_size=1 (2B)
            g1[1] = (int)((unsigned)DDIM << 16);             // tensor_dim0[15:0] @ [31:16]
            g1[2] = (int)(((unsigned)DDIM >> 16) | ((unsigned)(T_TOK & 0xFFFF) << 16)); // dim0 hi | dim1 lo
            g1[3] = (int)(((unsigned)T_TOK >> 16) | ((unsigned)DDIM << 16));            // dim1 hi | tile_dim0
            g1[4] = 16;                                      // tile_dim1 = #indices
            g1[5] = DDIM;                                    // tensor_dim0_stride lo32
            g1[6] = 0;
            g1[7] = 0;
            i32x4 g2, g3;
#pragma unroll
            for (int i = 0; i < 4; ++i) {
                int p0 = tk[half * 16 + 2 * i],     p1 = tk[half * 16 + 2 * i + 1];
                int p2 = tk[half * 16 + 8 + 2 * i], p3 = tk[half * 16 + 8 + 2 * i + 1];
                int t0 = (p0 >= 0) ? (p0 >> 1) : 0, t1 = (p1 >= 0) ? (p1 >> 1) : 0;
                int t2 = (p2 >= 0) ? (p2 >> 1) : 0, t3 = (p3 >= 0) ? (p3 >> 1) : 0;
                g2[i] = t0 | (t1 << 16);
                g3[i] = t2 | (t3 << 16);
            }
            i32x8 g4 = {0, 0, 0, 0, 0, 0, 0, 0};             // unused trailing group (clang-23 6-arg form)
            __builtin_amdgcn_tensor_load_to_lds(g0, g1, g2, g3, g4, 0);
        }
        __builtin_amdgcn_s_wait_tensorcnt(0);
    }
#else
    for (int i = tid; i < 32 * (DDIM / 8); i += 256) {       // 128 uint4 per row
        int row = i >> 7, off = i & 127;
        int packed = tk[row];
        int tok = (packed >= 0) ? (packed >> 1) : 0;
        ((uint4*)xrow)[row * (DDIM / 8) + off] =
            ((const uint4*)(xin + (size_t)tok * DDIM))[off];
    }
#endif
    __syncthreads();

    // ---- persistent output accumulators: 2 m-tiles x 8 n-tiles of D
    f32x8 yacc[2][8];
#pragma unroll
    for (int j = 0; j < 8; ++j) {
        float c0 = b2[e * DDIM + (wv * 8 + j) * 16 + nlo];
#pragma unroll
        for (int v = 0; v < 8; ++v) { yacc[0][j][v] = c0; yacc[1][j][v] = c0; }
    }

    const bf16* W1e = w1p + (size_t)e * (FDIM / 16) * (DDIM / 32) * 512;
    const bf16* W2e = w2p + (size_t)e * (DDIM / 16) * (FDIM / 32) * 512;

    for (int fb = 0; fb < FDIM; fb += FCH) {
        // ---- Phase A: h_chunk = gelu(x @ W1[:,fb:fb+FCH] + b1) -> hfrag (A layout)
#pragma unroll 1
        for (int jp = 0; jp < 4; ++jp) {
            int nf0 = fb + (wv * 8 + jp * 2) * 16;
            int nf1 = nf0 + 16;
            f32x8 h00, h01, h10, h11;
            float c00 = b1[e * FDIM + nf0 + nlo];
            float c01 = b1[e * FDIM + nf1 + nlo];
#pragma unroll
            for (int v = 0; v < 8; ++v) {
                h00[v] = c00; h10[v] = c00; h01[v] = c01; h11[v] = c01;
            }
            const bf16* bp0 = W1e + ((size_t)(nf0 >> 4) * (DDIM / 32)) * 512 + lane * 16;
            const bf16* bp1 = W1e + ((size_t)(nf1 >> 4) * (DDIM / 32)) * 512 + lane * 16;
            for (int kt = 0; kt < DDIM / 32; ++kt) {
                bf16x16 a0 = afrag_rows(xrow + (0 * 16 + nlo) * DDIM + kt * 32, hs);
                bf16x16 a1 = afrag_rows(xrow + (1 * 16 + nlo) * DDIM + kt * 32, hs);
                bf16x16 q0 = *(const bf16x16*)(bp0 + kt * 512);
                bf16x16 q1 = *(const bf16x16*)(bp1 + kt * 512);
                h00 = __builtin_amdgcn_wmma_f32_16x16x32_bf16(false, a0, false, q0, (short)0, h00, false, false);
                h10 = __builtin_amdgcn_wmma_f32_16x16x32_bf16(false, a1, false, q0, (short)0, h10, false, false);
                h01 = __builtin_amdgcn_wmma_f32_16x16x32_bf16(false, a0, false, q1, (short)0, h01, false, false);
                h11 = __builtin_amdgcn_wmma_f32_16x16x32_bf16(false, a1, false, q1, (short)0, h11, false, false);
            }
            // gelu -> store into hfrag (A-fragment layout), both m-tiles, both n-tiles
#pragma unroll
            for (int q = 0; q < 2; ++q) {
                int nfl = (nf0 - fb) + q * 16;
                int kt2 = nfl >> 5;
                int hsp = (nlo >> 3) & 1;
                int elp = (nlo & 7) + ((nfl >> 4) & 1) * 8;
#pragma unroll
                for (int v = 0; v < 8; ++v) {
                    int row = hs * 8 + v;
                    float g0v = gelu_f(q ? h01[v] : h00[v]);
                    float g1v = gelu_f(q ? h11[v] : h10[v]);
                    hfrag[(0 * (FCH / 32) + kt2) * 512 + (hsp * 16 + row) * 16 + elp] = (bf16)g0v;
                    hfrag[(1 * (FCH / 32) + kt2) * 512 + (hsp * 16 + row) * 16 + elp] = (bf16)g1v;
                }
            }
        }
        __syncthreads();

        // ---- Phase B: yacc += h_chunk @ W2[fb:fb+FCH, :]
#pragma unroll 1
        for (int kt = 0; kt < FCH / 32; ++kt) {
            bf16x16 a0 = *(const bf16x16*)(hfrag + (0 * (FCH / 32) + kt) * 512 + lane * 16);
            bf16x16 a1 = *(const bf16x16*)(hfrag + (1 * (FCH / 32) + kt) * 512 + lane * 16);
            int ktg = (fb >> 5) + kt;
#pragma unroll
            for (int j = 0; j < 8; ++j) {
                bf16x16 b = *(const bf16x16*)(
                    W2e + ((size_t)(wv * 8 + j) * (FDIM / 32) + ktg) * 512 + lane * 16);
                yacc[0][j] = __builtin_amdgcn_wmma_f32_16x16x32_bf16(false, a0, false, b, (short)0, yacc[0][j], false, false);
                yacc[1][j] = __builtin_amdgcn_wmma_f32_16x16x32_bf16(false, a1, false, b, (short)0, yacc[1][j], false, false);
            }
        }
        __syncthreads();
    }

    // ---- apply routing weight, store per-(token,slot) partials
#pragma unroll
    for (int mt = 0; mt < 2; ++mt) {
#pragma unroll
        for (int j = 0; j < 8; ++j) {
            int nd = (wv * 8 + j) * 16 + nlo;
#pragma unroll
            for (int v = 0; v < 8; ++v) {
                int row = mt * 16 + hs * 8 + v;
                int packed = tk[row];
                if (packed >= 0)
                    outp[(size_t)packed * DDIM + nd] = yacc[mt][j][v] * wt[row];
            }
        }
    }
}

// ---------------------------------------------------------------- combine
__global__ void k_combine(const float* __restrict__ p, float* __restrict__ out, int n) {
    int i = blockIdx.x * blockDim.x + threadIdx.x;
    if (i < n) {
        int t = i >> 10;            // DDIM == 1024
        int d = i & 1023;
        out[i] = p[(size_t)(2 * t) * DDIM + d] + p[(size_t)(2 * t + 1) * DDIM + d];
    }
}

// ---------------------------------------------------------------- launch

extern "C" void kernel_launch(void* const* d_in, const int* in_sizes, int n_in,
                              void* d_out, int out_size, void* d_ws, size_t ws_size,
                              hipStream_t stream) {
    const float* gate_inputs = (const float*)d_in[0];
    const float* inputs      = (const float*)d_in[1];
    const float* Wg          = (const float*)d_in[2];
    const float* bg          = (const float*)d_in[3];
    const float* w1          = (const float*)d_in[4];
    const float* b1          = (const float*)d_in[5];
    const float* w2          = (const float*)d_in[6];
    const float* b2          = (const float*)d_in[7];
    float* out = (float*)d_out;

    char* ws = (char*)d_ws;
    size_t off = 0;
    bf16* x_bf  = (bf16*)(ws + off); off += (size_t)T_TOK * DDIM * 2;        // 16 MB
    bf16* w1p   = (bf16*)(ws + off); off += (size_t)NEXP * DDIM * FDIM * 2;  // 64 MB packed
    bf16* w2p   = (bf16*)(ws + off); off += (size_t)NEXP * FDIM * DDIM * 2;  // 64 MB packed
    float* outp = (float*)(ws + off); off += (size_t)T_TOK * 2 * DDIM * 4;   // 64 MB
    int* counts = (int*)(ws + off);  off += 64;
    int* list   = (int*)(ws + off);  off += (size_t)NEXP * T_TOK * 4;
    float* wlist = (float*)(ws + off); off += (size_t)NEXP * T_TOK * 4;
    (void)ws_size; (void)in_sizes; (void)n_in; (void)out_size;

    k_zero_counts<<<1, 32, 0, stream>>>(counts);

    {   int n4 = T_TOK * DDIM / 4;
        k_convert_bf16x4<<<(n4 + 255) / 256, 256, 0, stream>>>((const float4*)inputs, x_bf, n4); }

    {   int waves = NEXP * (FDIM / 16) * (DDIM / 32);   // 65536
        k_pack_b<<<waves / 8, 256, 0, stream>>>(w1, w1p, DDIM, FDIM); }
    {   int waves = NEXP * (DDIM / 16) * (FDIM / 32);   // 65536
        k_pack_b<<<waves / 8, 256, 0, stream>>>(w2, w2p, FDIM, DDIM); }

    k_gate_topk<<<T_TOK / 8, 256, 0, stream>>>(gate_inputs, Wg, bg, counts, list, wlist);

    size_t smem = (size_t)32 * DDIM * 2            // x rows (64 KB)
                + (size_t)2 * (FCH / 32) * 512 * 2 // h-chunk fragments (64 KB)
                + 32 * sizeof(int) + 32 * sizeof(float);
    dim3 grid(T_TOK / 32, NEXP);
    k_moe_ffn<<<grid, 256, smem, stream>>>(x_bf, w1p, b1, w2p, b2,
                                           counts, list, wlist, outp);

    k_combine<<<(T_TOK * DDIM + 255) / 256, 256, 0, stream>>>(outp, out, T_TOK * DDIM);
}
